// MPNN3_2920577761286
// MI455X (gfx1250) — matrix operand order repaired and checked by reading, not compiled, not run on hardware
//
#include <hip/hip_runtime.h>

#define N_NODES 10000
#define N_EDGES 50000
#define HDIM 64
#define NB 8

typedef __attribute__((ext_vector_type(16))) _Float16 v16h;
typedef __attribute__((ext_vector_type(8)))  float    v8f;

// ---------------------------------------------------------------------------
// WMMA helpers (CDNA5 wave32, 16x16x32 f16 -> f32)
// ---------------------------------------------------------------------------
__device__ inline v8f wmma_f16(v16h a, v16h b, v8f c) {
  return __builtin_amdgcn_wmma_f32_16x16x32_f16(
      /*neg_a=*/false, a, /*neg_b=*/false, b,
      /*c_mod=*/(short)0, c, /*reuse_a=*/false, /*reuse_b=*/false);
}

// A fragment: 16x32 (MxK) f16, source row-major with stride 64 (LDS tile).
// lanes 0-15: M=lane, elems = K[kt*32 + 0..7], K[kt*32 + 16..23]
// lanes16-31: M=lane-16, elems = K[kt*32 + 8..15], K[kt*32 + 24..31]
__device__ inline v16h load_a_frag_lds(const _Float16* sA, int lane, int kt) {
  int m = lane & 15;
  int ko = kt * 32 + ((lane >> 4) << 3);
  const _Float16* row = sA + m * 64;
  v16h a;
#pragma unroll
  for (int j = 0; j < 8; ++j) {
    a[j]     = row[ko + j];
    a[j + 8] = row[ko + 16 + j];
  }
  return a;
}

// B fragment: 32x16 (KxN) f16 from a row-major [64 x ld] weight matrix.
// lane = column n; lanes 0-15: K 0-7,16-23; lanes 16-31: K 8-15,24-31 of same columns.
__device__ inline v16h load_b_frag(const _Float16* W, int ld, int colBase, int lane, int kt) {
  int n  = lane & 15;
  int kb = kt * 32 + ((lane >> 4) << 3);
  v16h b;
#pragma unroll
  for (int j = 0; j < 16; ++j) b[j] = W[(size_t)(kb + (j & 7) + ((j >> 3) << 4)) * ld + colBase + n];
  return b;
}

__device__ inline float sigmoidf_(float x) { return 1.0f / (1.0f + __expf(-x)); }

__device__ inline void atomicMaxF(float* addr, float val) {
  unsigned int* ua = (unsigned int*)addr;
  unsigned int old = __float_as_uint(*addr);
  while (__uint_as_float(old) < val) {
    unsigned int assumed = old;
    old = atomicCAS(ua, assumed, __float_as_uint(val));
    if (old == assumed) break;
  }
}

// ---------------------------------------------------------------------------
// Small utility kernels
// ---------------------------------------------------------------------------
__global__ void zero_kernel(float* p, int n) {
  int i = blockIdx.x * blockDim.x + threadIdx.x;
  if (i < n) p[i] = 0.0f;
}
__global__ void fill_kernel(float* p, int n, float v) {
  int i = blockIdx.x * blockDim.x + threadIdx.x;
  if (i < n) p[i] = v;
}

// Build f16 weight layouts:
//  Wcat [64 x 320]: col = plane*64 + o; plane 0..3 = Wk (edge_mlp_w row k), plane 4 = bias mat
//  rootw16 [64 x 64] (direct), wih_t/whh_t [64 x 192] = gru_w_*^T
__global__ void prep_weights_kernel(const float* __restrict__ emw, const float* __restrict__ emb,
                                    const float* __restrict__ rootw,
                                    const float* __restrict__ wih, const float* __restrict__ whh,
                                    _Float16* Wcat, _Float16* rootw16,
                                    _Float16* wih_t, _Float16* whh_t) {
  int idx = blockIdx.x * blockDim.x + threadIdx.x;
  if (idx < 64 * 320) {
    int i = idx / 320, c = idx % 320;
    int p = c / 64, o = c % 64;
    float v = (p < 4) ? emw[p * 4096 + i * 64 + o] : emb[i * 64 + o];
    Wcat[idx] = (_Float16)v;
  }
  if (idx < 64 * 64) rootw16[idx] = (_Float16)rootw[idx];
  if (idx < 64 * 192) {
    int i = idx / 192, j = idx % 192;
    wih_t[idx] = (_Float16)wih[j * 64 + i];
    whh_t[idx] = (_Float16)whh[j * 64 + i];
  }
}

// h = relu(x @ lin0_w + lin0_b)  (NF=16 -> trivial VALU loop), dual f32/f16 output
__global__ void lin0_kernel(const float* __restrict__ x, const float* __restrict__ w,
                            const float* __restrict__ b, float* __restrict__ h32,
                            _Float16* __restrict__ h16) {
  int idx = blockIdx.x * blockDim.x + threadIdx.x;
  if (idx >= N_NODES * 64) return;
  int n = idx >> 6, o = idx & 63;
  const float* xr = x + n * 16;
  float acc = b[o];
#pragma unroll
  for (int i = 0; i < 16; ++i) acc += xr[i] * w[i * 64 + o];
  acc = fmaxf(acc, 0.0f);
  h32[idx] = acc;
  h16[idx] = (_Float16)acc;
}

// ---------------------------------------------------------------------------
// NNConv message kernel: T = gather(h[src]) @ Wcat; msg = ea-combine; scatter-add.
// Block = 128 threads = 4 waves; wave w owns output cols [16w,16w+16) of all 5 planes.
// ---------------------------------------------------------------------------
__global__ __launch_bounds__(128) void msg_kernel(const _Float16* __restrict__ h16,
                                                  const float* __restrict__ ea,
                                                  const int* __restrict__ srcI,
                                                  const int* __restrict__ dstI,
                                                  const _Float16* __restrict__ Wcat,
                                                  float* __restrict__ agg, int nTiles) {
  __shared__ __align__(16) _Float16 sA[16 * 64];
  __shared__ float sEA[16 * 4];
  __shared__ int sDst[16];
  int tid = threadIdx.x;
  int lane = tid & 31, wv = tid >> 5;
  int co = wv * 16;

  // Loop-invariant B fragments: 5 planes x 2 K-tiles, kept in VGPRs.
  v16h bf[5][2];
#pragma unroll
  for (int p = 0; p < 5; ++p)
#pragma unroll
    for (int kt = 0; kt < 2; ++kt)
      bf[p][kt] = load_b_frag(Wcat, 320, p * 64 + co, lane, kt);

  for (int tile = blockIdx.x; tile < nTiles; tile += gridDim.x) {
    int e0 = tile * 16;
    { // stage: gather 16 source rows (f16, 128B each) + edge attrs + dst ids
      int er = tid >> 3, seg = tid & 7;
      int s = srcI[e0 + er];
      ((float4*)sA)[er * 8 + seg] = ((const float4*)(h16 + (size_t)s * 64))[seg];
      if (tid < 64) sEA[tid] = ea[(size_t)e0 * 4 + tid];
      if (tid >= 64 && tid < 80) sDst[tid - 64] = dstI[e0 + tid - 64];
    }
    __syncthreads();

    v16h a0 = load_a_frag_lds(sA, lane, 0);
    v16h a1 = load_a_frag_lds(sA, lane, 1);
    v8f c0 = {}, c1 = {}, c2 = {}, c3 = {}, c4 = {};
    c0 = wmma_f16(a0, bf[0][0], c0); c0 = wmma_f16(a1, bf[0][1], c0);
    c1 = wmma_f16(a0, bf[1][0], c1); c1 = wmma_f16(a1, bf[1][1], c1);
    c2 = wmma_f16(a0, bf[2][0], c2); c2 = wmma_f16(a1, bf[2][1], c2);
    c3 = wmma_f16(a0, bf[3][0], c3); c3 = wmma_f16(a1, bf[3][1], c3);
    c4 = wmma_f16(a0, bf[4][0], c4); c4 = wmma_f16(a1, bf[4][1], c4);

    int n = lane & 15, mo = (lane >> 4) * 8;
#pragma unroll
    for (int r = 0; r < 8; ++r) {
      int m = mo + r;
      float v = c4[r] + sEA[m * 4 + 0] * c0[r] + sEA[m * 4 + 1] * c1[r]
                      + sEA[m * 4 + 2] * c2[r] + sEA[m * 4 + 3] * c3[r];
      atomicAdd(agg + (size_t)sDst[m] * 64 + co + n, v);
    }
    __syncthreads();
  }
}

// ---------------------------------------------------------------------------
// Node update: m = relu(agg + h@root_w + conv_b); GRU step; in-place h update.
// Block = 128 threads = 4 waves; 16-node tiles (contiguous rows).
// ---------------------------------------------------------------------------
__global__ __launch_bounds__(128) void node_kernel(float* __restrict__ h32,
                                                   _Float16* __restrict__ h16,
                                                   const float* __restrict__ agg,
                                                   const _Float16* __restrict__ rw16,
                                                   const _Float16* __restrict__ wih_t,
                                                   const _Float16* __restrict__ whh_t,
                                                   const float* __restrict__ convb,
                                                   const float* __restrict__ bih,
                                                   const float* __restrict__ bhh, int nTiles) {
  __shared__ __align__(16) _Float16 sH[16 * 64];
  __shared__ __align__(16) _Float16 sM[16 * 64];
  __shared__ float sGI[16 * 192];
  __shared__ float sGH[16 * 192];
  int tid = threadIdx.x, lane = tid & 31, wv = tid >> 5;

  v16h bR[2], bI[3][2], bH[3][2];
#pragma unroll
  for (int kt = 0; kt < 2; ++kt) bR[kt] = load_b_frag(rw16, 64, wv * 16, lane, kt);
#pragma unroll
  for (int jt = 0; jt < 3; ++jt)
#pragma unroll
    for (int kt = 0; kt < 2; ++kt) {
      bI[jt][kt] = load_b_frag(wih_t, 192, wv * 48 + jt * 16, lane, kt);
      bH[jt][kt] = load_b_frag(whh_t, 192, wv * 48 + jt * 16, lane, kt);
    }

  for (int tile = blockIdx.x; tile < nTiles; tile += gridDim.x) {
    int n0 = tile * 16;
    { // stage h tile (contiguous rows)
      int er = tid >> 3, seg = tid & 7;
      ((float4*)sH)[er * 8 + seg] = ((const float4*)(h16 + (size_t)(n0 + er) * 64))[seg];
    }
    __syncthreads();
    v16h aH0 = load_a_frag_lds(sH, lane, 0);
    v16h aH1 = load_a_frag_lds(sH, lane, 1);

    { // stage 1: R = h @ root_w ; m = relu(agg + R + conv_b) -> sM (f16)
      v8f cr = {};
      cr = wmma_f16(aH0, bR[0], cr);
      cr = wmma_f16(aH1, bR[1], cr);
      int n = lane & 15, mo = (lane >> 4) * 8, o = wv * 16 + n;
      float cb = convb[o];
#pragma unroll
      for (int r = 0; r < 8; ++r) {
        int m = mo + r;
        float mv = fmaxf(agg[(size_t)(n0 + m) * 64 + o] + cr[r] + cb, 0.0f);
        sM[m * 64 + o] = (_Float16)mv;
      }
    }
    __syncthreads();

    v16h aM0 = load_a_frag_lds(sM, lane, 0);
    v16h aM1 = load_a_frag_lds(sM, lane, 1);
    { // stage 2: gi = m@Wih^T + b_ih, gh = h@Whh^T + b_hh -> LDS
      int n = lane & 15, mo = (lane >> 4) * 8;
#pragma unroll
      for (int jt = 0; jt < 3; ++jt) {
        int col = wv * 48 + jt * 16 + n;
        v8f ci = {}, ch = {};
        ci = wmma_f16(aM0, bI[jt][0], ci);
        ci = wmma_f16(aM1, bI[jt][1], ci);
        ch = wmma_f16(aH0, bH[jt][0], ch);
        ch = wmma_f16(aH1, bH[jt][1], ch);
        float bi = bih[col], bh = bhh[col];
#pragma unroll
        for (int r = 0; r < 8; ++r) {
          int m = mo + r;
          sGI[m * 192 + col] = ci[r] + bi;
          sGH[m * 192 + col] = ch[r] + bh;
        }
      }
    }
    __syncthreads();

    // GRU elementwise (torch gate order r,z,n), in-place h update
    for (int idx = tid; idx < 1024; idx += 128) {
      int m = idx >> 6, o = idx & 63;
      const float* gi = sGI + m * 192;
      const float* gh = sGH + m * 192;
      float r = sigmoidf_(gi[o] + gh[o]);
      float z = sigmoidf_(gi[64 + o] + gh[64 + o]);
      float nn = tanhf(gi[128 + o] + r * gh[128 + o]);
      size_t g = (size_t)(n0 + m) * 64 + o;
      float hold = h32[g];
      float hn = (1.0f - z) * nn + z * hold;
      h32[g] = hn;
      h16[g] = (_Float16)hn;
    }
    __syncthreads();
  }
}

// ---------------------------------------------------------------------------
// Set2Set (B = 8) kernels
// ---------------------------------------------------------------------------
__global__ void lstm_kernel(const float* __restrict__ qstar, float* __restrict__ hl,
                            float* __restrict__ cl, const float* __restrict__ wih,
                            const float* __restrict__ whh, const float* __restrict__ bih,
                            const float* __restrict__ bhh) {
  __shared__ float g[NB * 256];
  int j = threadIdx.x;  // 256 threads: one gate column each
  for (int b = 0; b < NB; ++b) {
    float acc = bih[j] + bhh[j];
    const float* q = qstar + b * 128;
    const float* wi = wih + (size_t)j * 128;
    for (int i = 0; i < 128; ++i) acc += q[i] * wi[i];
    const float* hrow = hl + b * 64;
    const float* wh = whh + (size_t)j * 64;
    for (int i = 0; i < 64; ++i) acc += hrow[i] * wh[i];
    g[b * 256 + j] = acc;
  }
  __syncthreads();
  for (int idx = threadIdx.x; idx < NB * 64; idx += 256) {
    int b = idx >> 6, o = idx & 63;
    const float* gb = g + b * 256;
    float ig = sigmoidf_(gb[o]);
    float fg = sigmoidf_(gb[64 + o]);
    float gg = tanhf(gb[128 + o]);
    float og = sigmoidf_(gb[192 + o]);
    float c = fg * cl[idx] + ig * gg;
    cl[idx] = c;
    hl[idx] = og * tanhf(c);
  }
}

__global__ void dot_max_kernel(const float* __restrict__ h, const float* __restrict__ hl,
                               const int* __restrict__ batch, float* __restrict__ e,
                               float* __restrict__ maxb, int N) {
  __shared__ float sMax[NB];
  if (threadIdx.x < NB) sMax[threadIdx.x] = -3.0e38f;
  __syncthreads();
  int n = blockIdx.x * blockDim.x + threadIdx.x;
  if (n < N) {
    int b = batch[n];
    const float* hr = h + (size_t)n * 64;
    const float* q = hl + b * 64;
    float acc = 0.0f;
    for (int i = 0; i < 64; ++i) acc += hr[i] * q[i];
    e[n] = acc;
    atomicMaxF(&sMax[b], acc);
  }
  __syncthreads();
  if (threadIdx.x < NB) atomicMaxF(&maxb[threadIdx.x], sMax[threadIdx.x]);
}

__global__ void exp_sum_kernel(const float* __restrict__ e, const float* __restrict__ maxb,
                               const int* __restrict__ batch, float* __restrict__ p,
                               float* __restrict__ sumb, int N) {
  __shared__ float sS[NB];
  if (threadIdx.x < NB) sS[threadIdx.x] = 0.0f;
  __syncthreads();
  int n = blockIdx.x * blockDim.x + threadIdx.x;
  if (n < N) {
    int b = batch[n];
    float v = __expf(e[n] - maxb[b]);
    p[n] = v;
    atomicAdd(&sS[b], v);
  }
  __syncthreads();
  if (threadIdx.x < NB) atomicAdd(&sumb[threadIdx.x], sS[threadIdx.x]);
}

__global__ void rvec_kernel(const float* __restrict__ h, const float* __restrict__ p,
                            const float* __restrict__ sumb, const int* __restrict__ batch,
                            float* __restrict__ rvec, int N) {
  __shared__ float sR[NB * 64];
  for (int i = threadIdx.x; i < NB * 64; i += blockDim.x) sR[i] = 0.0f;
  __syncthreads();
  int n = blockIdx.x * blockDim.x + threadIdx.x;
  if (n < N) {
    int b = batch[n];
    float a = p[n] / sumb[b];
    const float* hr = h + (size_t)n * 64;
    for (int o = 0; o < 64; ++o) atomicAdd(&sR[b * 64 + o], a * hr[o]);
  }
  __syncthreads();
  for (int i = threadIdx.x; i < NB * 64; i += blockDim.x) atomicAdd(&rvec[i], sR[i]);
}

__global__ void qstar_kernel(const float* __restrict__ hl, const float* __restrict__ rvec,
                             float* __restrict__ qstar) {
  int idx = blockIdx.x * blockDim.x + threadIdx.x;
  if (idx < NB * 128) {
    int b = idx >> 7, j = idx & 127;
    qstar[idx] = (j < 64) ? hl[b * 64 + j] : rvec[b * 64 + j - 64];
  }
}

__global__ void final_kernel(const float* __restrict__ qstar, const float* __restrict__ w1,
                             const float* __restrict__ b1, const float* __restrict__ w2,
                             const float* __restrict__ b2, float* __restrict__ y) {
  __shared__ float hid[NB * 64];
  int t = threadIdx.x;  // 64 threads
  for (int b = 0; b < NB; ++b) {
    float acc = b1[t];
    for (int i = 0; i < 128; ++i) acc += qstar[b * 128 + i] * w1[i * 64 + t];
    hid[b * 64 + t] = fmaxf(acc, 0.0f);
  }
  __syncthreads();
  if (t < NB) {
    float acc = b2[0];
    for (int o = 0; o < 64; ++o) acc += hid[t * 64 + o] * w2[o];
    y[t] = acc;
  }
}

// ---------------------------------------------------------------------------
// Orchestration
// ---------------------------------------------------------------------------
extern "C" void kernel_launch(void* const* d_in, const int* in_sizes, int n_in,
                              void* d_out, int out_size, void* d_ws, size_t ws_size,
                              hipStream_t stream) {
  (void)in_sizes; (void)n_in; (void)out_size; (void)ws_size;
  const float* x         = (const float*)d_in[0];
  const float* edge_attr = (const float*)d_in[1];
  const int*   edge_idx  = (const int*)d_in[2];
  const int*   batch     = (const int*)d_in[3];
  const float* lin0_w    = (const float*)d_in[4];
  const float* lin0_b    = (const float*)d_in[5];
  const float* edge_mlp_w= (const float*)d_in[6];
  const float* edge_mlp_b= (const float*)d_in[7];
  const float* root_w    = (const float*)d_in[8];
  const float* conv_b    = (const float*)d_in[9];
  const float* gru_w_ih  = (const float*)d_in[10];
  const float* gru_w_hh  = (const float*)d_in[11];
  const float* gru_b_ih  = (const float*)d_in[12];
  const float* gru_b_hh  = (const float*)d_in[13];
  const float* lstm_w_ih = (const float*)d_in[14];
  const float* lstm_w_hh = (const float*)d_in[15];
  const float* lstm_b_ih = (const float*)d_in[16];
  const float* lstm_b_hh = (const float*)d_in[17];
  const float* lin1_w    = (const float*)d_in[18];
  const float* lin1_b    = (const float*)d_in[19];
  const float* lin2_w    = (const float*)d_in[20];
  const float* lin2_b    = (const float*)d_in[21];
  const int* srcI = edge_idx;
  const int* dstI = edge_idx + N_EDGES;

  char* ws = (char*)d_ws;
  size_t off = 0;
  auto alloc = [&](size_t bytes) -> void* {
    void* p = ws + off;
    off += (bytes + 255) & ~(size_t)255;
    return p;
  };
  float*     h32    = (float*)alloc((size_t)N_NODES * 64 * 4);
  _Float16*  h16    = (_Float16*)alloc((size_t)N_NODES * 64 * 2);
  float*     agg    = (float*)alloc((size_t)N_NODES * 64 * 4);
  _Float16*  Wcat   = (_Float16*)alloc(64 * 320 * 2);
  _Float16*  rw16   = (_Float16*)alloc(64 * 64 * 2);
  _Float16*  wih_t  = (_Float16*)alloc(64 * 192 * 2);
  _Float16*  whh_t  = (_Float16*)alloc(64 * 192 * 2);
  float*     e_arr  = (float*)alloc(N_NODES * 4);
  float*     p_arr  = (float*)alloc(N_NODES * 4);
  float*     maxb   = (float*)alloc(NB * 4);
  float*     sumb   = (float*)alloc(NB * 4);
  float*     hl     = (float*)alloc(NB * 64 * 4);
  float*     cl     = (float*)alloc(NB * 64 * 4);
  float*     rvec   = (float*)alloc(NB * 64 * 4);
  float*     qstar  = (float*)alloc(NB * 128 * 4);

  // Weight prep + input embedding
  prep_weights_kernel<<<(64 * 320 + 255) / 256, 256, 0, stream>>>(
      edge_mlp_w, edge_mlp_b, root_w, gru_w_ih, gru_w_hh, Wcat, rw16, wih_t, whh_t);
  lin0_kernel<<<(N_NODES * 64 + 255) / 256, 256, 0, stream>>>(x, lin0_w, lin0_b, h32, h16);

  // 3 message-passing rounds
  const int edgeTiles = N_EDGES / 16;   // 3125
  const int nodeTiles = N_NODES / 16;   // 625
  for (int it = 0; it < 3; ++it) {
    zero_kernel<<<(N_NODES * 64 + 255) / 256, 256, 0, stream>>>(agg, N_NODES * 64);
    msg_kernel<<<512, 128, 0, stream>>>(h16, edge_attr, srcI, dstI, Wcat, agg, edgeTiles);
    node_kernel<<<256, 128, 0, stream>>>(h32, h16, agg, rw16, wih_t, whh_t,
                                         conv_b, gru_b_ih, gru_b_hh, nodeTiles);
  }

  // Set2Set init
  zero_kernel<<<2, 256, 0, stream>>>(hl, NB * 64);
  zero_kernel<<<2, 256, 0, stream>>>(cl, NB * 64);
  zero_kernel<<<4, 256, 0, stream>>>(qstar, NB * 128);

  const int nodeBlocks = (N_NODES + 255) / 256;
  for (int s = 0; s < 3; ++s) {
    lstm_kernel<<<1, 256, 0, stream>>>(qstar, hl, cl, lstm_w_ih, lstm_w_hh, lstm_b_ih, lstm_b_hh);
    fill_kernel<<<1, NB, 0, stream>>>(maxb, NB, -3.0e38f);
    zero_kernel<<<1, NB, 0, stream>>>(sumb, NB);
    zero_kernel<<<2, 256, 0, stream>>>(rvec, NB * 64);
    dot_max_kernel<<<nodeBlocks, 256, 0, stream>>>(h32, hl, batch, e_arr, maxb, N_NODES);
    exp_sum_kernel<<<nodeBlocks, 256, 0, stream>>>(e_arr, maxb, batch, p_arr, sumb, N_NODES);
    rvec_kernel<<<nodeBlocks, 256, 0, stream>>>(h32, p_arr, sumb, batch, rvec, N_NODES);
    qstar_kernel<<<4, 256, 0, stream>>>(hl, rvec, qstar);
  }

  final_kernel<<<1, 64, 0, stream>>>(qstar, lin1_w, lin1_b, lin2_w, lin2_b, (float*)d_out);
}
